// ResnetEncoderMatching_15066745274575
// MI455X (gfx1250) — compile-verified, matrix operand order checked by reading, not run on hardware
//
#include <hip/hip_runtime.h>

// Problem constants (match reference).
#define B_ 2
#define F_ 2
#define C_ 64
#define H_ 32
#define W_ 104
#define D_ 96
#define EPS_ 1e-7f
#define XB_ 7              // ceil(W/16) pixel groups per row

typedef __attribute__((ext_vector_type(16))) _Float16 v16h;
typedef __attribute__((ext_vector_type(8)))  float    v8f;
typedef __attribute__((ext_vector_type(4)))  float    v4f;

// Workspace layout (float elements)
#define LFT_OFF   0
#define LFT_N     (B_*F_*H_*W_*C_)          // lookup feats, channel-last
#define CURT_OFF  (LFT_OFF + LFT_N)
#define CURT_N    (B_*H_*W_*C_)             // current feats, channel-last
#define PM_OFF    (CURT_OFF + CURT_N)
#define PM_N      (B_*F_*12)                // (K @ pose)[:3,:]
#define VALID_OFF (PM_OFF + PM_N)           // B_*F_ valid flags

__device__ __forceinline__ int iclamp(int v, int lo, int hi) {
    return v < lo ? lo : (v > hi ? hi : v);
}

// ---- Prep: P = (K @ pose)[:3,:], valid = (sum(pose) != 0) ----
__global__ void prep_kernel(const float* __restrict__ K,
                            const float* __restrict__ poses,
                            float* __restrict__ ws) {
    int t = threadIdx.x;
    if (t >= B_ * F_) return;
    int b = t / F_;
    const float* Kb = K + b * 16;
    const float* Pp = poses + t * 16;
    float s = 0.f;
    for (int i = 0; i < 16; ++i) s += Pp[i];
    ws[VALID_OFF + t] = (s != 0.f) ? 1.f : 0.f;
    float* M = ws + PM_OFF + t * 12;
    for (int i = 0; i < 3; ++i)
        for (int j = 0; j < 4; ++j) {
            float acc = 0.f;
            for (int k = 0; k < 4; ++k) acc += Kb[i*4 + k] * Pp[k*4 + j];
            M[i*4 + j] = acc;
        }
}

// ---- Transpose lookup feats [B,F,C,H,W] -> [B,F,H,W,C] ----
__global__ void transpose_lookup(const float* __restrict__ lf, float* __restrict__ ws) {
    int tid = blockIdx.x * blockDim.x + threadIdx.x;   // B*F*C*H*W, exact
    int x = tid % W_; int r = tid / W_;
    int y = r % H_;  r /= H_;
    int c = r % C_;  r /= C_;                          // r = b*F+f
    ws[LFT_OFF + (((size_t)r * H_ + y) * W_ + x) * C_ + c] = lf[tid];
}

// ---- Transpose current feats [B,C,H,W] -> [B,H,W,C] ----
__global__ void transpose_cur(const float* __restrict__ cf, float* __restrict__ ws) {
    int tid = blockIdx.x * blockDim.x + threadIdx.x;   // B*C*H*W, exact
    int x = tid % W_; int r = tid / W_;
    int y = r % H_;  r /= H_;
    int c = r % C_;  r /= C_;                          // r = b
    ws[CURT_OFF + (((size_t)r * H_ + y) * W_ + x) * C_ + c] = cf[tid];
}

// ---- Main cost-volume kernel ----
// One wave = 16 pixels (contiguous x) x 2 channel-halves.
// lane L: pixel p = L&15, channels [32*(L>>4), +32).
// Channel reduction over 64 channels done with v_wmma_f32_16x16x32_f16
// against an all-ones B matrix (fp32 accumulate; only pre-reduced
// partials pass through f16; mask folded in before the WMMA so the
// count = (diffs > 0) semantics survive exactly).
__global__ void __launch_bounds__(256)
cost_volume_kernel(const float* __restrict__ invK,
                   const float* __restrict__ depth_bins,
                   const float* __restrict__ ws,
                   float* __restrict__ out) {
    const int lane = threadIdx.x & 31;
    int gw = (blockIdx.x * blockDim.x + threadIdx.x) >> 5;  // global wave id
    int xb = gw % XB_; gw /= XB_;
    int y  = gw % H_;  gw /= H_;
    int dd = gw % D_;  gw /= D_;
    int b  = gw;

    const int p   = lane & 15;          // pixel in group
    const int hh  = lane >> 4;          // channel half
    const int ch0 = hh * 32;
    const int x   = xb * 16 + p;
    const float xf = (float)x, yf = (float)y;

    const float* iK = invK + b * 16;
    const float camx = iK[0]*xf + iK[1]*yf + iK[2];
    const float camy = iK[4]*xf + iK[5]*yf + iK[6];
    const float camz = iK[8]*xf + iK[9]*yf + iK[10];
    const float depth = depth_bins[dd];

    const bool interior = (x >= 2) && (x <= W_ - 3) && (y >= 2) && (y <= H_ - 3);

    const float* lfT  = ws + LFT_OFF;
    const float* curT = ws + CURT_OFF;
    const float* Pm   = ws + PM_OFF;
    const float* vld  = ws + VALID_OFF;

    // Current-frame features: independent of f and d -> load once, keep in VGPRs.
    const int xs = (x < W_) ? x : (W_ - 1);
    const float* curp = curT + (((size_t)b * H_ + y) * W_ + xs) * C_ + ch0;
    v4f curv[8];
#pragma unroll
    for (int j = 0; j < 8; ++j) curv[j] = *(const v4f*)(curp + 4*j);

    v16h Bones;
#pragma unroll
    for (int i = 0; i < 16; ++i) Bones[i] = (_Float16)1.0f;

    float accc[8], cnt[8];
#pragma unroll
    for (int r = 0; r < 8; ++r) { accc[r] = 0.f; cnt[r] = 0.f; }

    for (int f = 0; f < F_; ++f) {
        const float* M = Pm + (b * F_ + f) * 12;
        float cx = depth * (M[0]*camx + M[1]*camy + M[2]*camz)  + M[3];
        float cy = depth * (M[4]*camx + M[5]*camy + M[6]*camz)  + M[7];
        float cz = depth * (M[8]*camx + M[9]*camy + M[10]*camz) + M[11];
        // Raw v_rcp_f32 (~1 ulp) instead of the full IEEE divide sequence:
        // coordinate error ~1e-5 px, negligible vs bilinear tolerance.
        float iz = __builtin_amdgcn_rcpf(cz + EPS_);
        float gx = cx * iz, gy = cy * iz;
        bool edge = (gx >= 2.0f) && (gx <= (float)(W_ - 2)) &&
                    (gy >= 2.0f) && (gy <= (float)(H_ - 2));
        float maskv = (edge && interior) ? vld[b * F_ + f] : 0.f;

        float fx0 = floorf(gx), fy0 = floorf(gy);
        float wx1 = gx - fx0, wy1 = gy - fy0;
        float wx0 = 1.f - wx1, wy0 = 1.f - wy1;
        int x0i = (int)fx0, y0i = (int)fy0;
        int xc0 = iclamp(x0i,     0, W_ - 1), xc1 = iclamp(x0i + 1, 0, W_ - 1);
        int yc0 = iclamp(y0i,     0, H_ - 1), yc1 = iclamp(y0i + 1, 0, H_ - 1);

        const float* img = lfT + (size_t)(b * F_ + f) * H_ * W_ * C_;
        const float* p00 = img + ((size_t)yc0 * W_ + xc0) * C_ + ch0;
        const float* p01 = img + ((size_t)yc0 * W_ + xc1) * C_ + ch0;
        const float* p10 = img + ((size_t)yc1 * W_ + xc0) * C_ + ch0;
        const float* p11 = img + ((size_t)yc1 * W_ + xc1) * C_ + ch0;
        float w00 = wy0*wx0, w01 = wy0*wx1, w10 = wy1*wx0, w11 = wy1*wx1;

        v4f acc = {0.f, 0.f, 0.f, 0.f};
#pragma unroll
        for (int j = 0; j < 8; ++j) {
            v4f a00 = *(const v4f*)(p00 + 4*j);
            v4f a01 = *(const v4f*)(p01 + 4*j);
            v4f a10 = *(const v4f*)(p10 + 4*j);
            v4f a11 = *(const v4f*)(p11 + 4*j);
            v4f t = a00*w00 + a01*w01 + a10*w10 + a11*w11 - curv[j];
            acc.x += __builtin_fabsf(t.x);
            acc.y += __builtin_fabsf(t.y);
            acc.z += __builtin_fabsf(t.z);
            acc.w += __builtin_fabsf(t.w);
        }

        // Fold mask + 1/C into the partials, convert to f16, reduce via WMMA.
        float scale = maskv * (1.0f / (float)C_);
        v16h A = {};
        A[0] = (_Float16)(acc.x * scale);
        A[1] = (_Float16)(acc.y * scale);
        A[2] = (_Float16)(acc.z * scale);
        A[3] = (_Float16)(acc.w * scale);
        v8f Cz = {};
        // D[p][n] = sum of all 32 f16 slots held by lanes p and p+16  (fp32 acc)
        v8f Dv = __builtin_amdgcn_wmma_f32_16x16x32_f16(
            /*neg_a=*/false, A, /*neg_b=*/false, Bones,
            /*c_mod=*/(short)0, Cz, /*reuse_a=*/false, /*reuse_b=*/false);
#pragma unroll
        for (int r = 0; r < 8; ++r) {
            float dv = Dv[r];           // lanes 0-15: pixel r; lanes 16-31: pixel r+8
            accc[r] += dv;
            cnt[r]  += (dv > 0.f) ? 1.f : 0.f;
        }
    }

    // lanes 0 and 16 each own 8 consecutive pixels' results (column 0 of D).
    if (p == 0) {
        size_t row = (((size_t)b * D_ + dd) * H_ + y) * W_;
        int base = xb * 16 + hh * 8;
#pragma unroll
        for (int r = 0; r < 8; ++r) {
            int px = base + r;
            if (px < W_) out[row + px] = accc[r] / (cnt[r] + EPS_);
        }
    }
}

// ---- Replace zero (missing) entries with per-(b,y,x) max over depth ----
__global__ void fix_missing(float* __restrict__ out) {
    int tid = blockIdx.x * blockDim.x + threadIdx.x;   // B*H*W
    if (tid >= B_ * H_ * W_) return;
    int x = tid % W_; int r = tid / W_;
    int y = r % H_;  int b = r / H_;
    size_t base = (((size_t)b * D_) * H_ + y) * W_ + x;
    const size_t stride = (size_t)H_ * W_;
    float m = -1e30f;
    for (int d = 0; d < D_; ++d) m = fmaxf(m, out[base + d * stride]);
    for (int d = 0; d < D_; ++d) {
        size_t o = base + d * stride;
        if (out[o] == 0.f) out[o] = m;
    }
}

extern "C" void kernel_launch(void* const* d_in, const int* in_sizes, int n_in,
                              void* d_out, int out_size, void* d_ws, size_t ws_size,
                              hipStream_t stream) {
    const float* cur   = (const float*)d_in[0];  // current_feats [B,C,H,W]
    const float* lf    = (const float*)d_in[1];  // lookup_feats  [B,F,C,H,W]
    const float* poses = (const float*)d_in[2];  // [B,F,4,4]
    const float* K     = (const float*)d_in[3];  // [B,4,4]
    const float* invK  = (const float*)d_in[4];  // [B,4,4]
    const float* dbins = (const float*)d_in[5];  // [D]
    float* out = (float*)d_out;                  // [B,D,H,W]
    float* ws  = (float*)d_ws;

    prep_kernel<<<1, 64, 0, stream>>>(K, poses, ws);
    transpose_lookup<<<(B_*F_*C_*H_*W_) / 256, 256, 0, stream>>>(lf, ws);
    transpose_cur<<<(B_*C_*H_*W_) / 256, 256, 0, stream>>>(cur, ws);

    int waves  = B_ * D_ * H_ * XB_;             // 43008, divisible by 8
    cost_volume_kernel<<<waves / 8, 256, 0, stream>>>(invK, dbins, ws, out);

    fix_missing<<<(B_*H_*W_ + 255) / 256, 256, 0, stream>>>(out);
}
